// TransformerNet_64141041598757
// MI455X (gfx1250) — matrix operand order repaired and checked by reading, not compiled, not run on hardware
//
#include <hip/hip_runtime.h>
#include <math.h>

// ---------------------------------------------------------------------------
// Problem constants (match reference)
// ---------------------------------------------------------------------------
#define BATCH 128
#define LSEQ  128
#define JDIM  64
#define EDIM  128
#define SEQ   129          // L + 1 (id token prepended)
#define HEADS 2
#define HD    64           // EDIM / HEADS
#define NCAT  1000
#define LN_EPS 1e-5f

typedef __attribute__((ext_vector_type(16))) _Float16 v16h;
typedef __attribute__((ext_vector_type(8)))  float    v8f;
typedef __attribute__((ext_vector_type(4)))  unsigned int u32x4;
typedef __attribute__((ext_vector_type(8)))  int      i32x8;
typedef __attribute__((ext_vector_type(4)))  int      i32x4;

// ---------------------------------------------------------------------------
// WMMA fragment swizzles (CDNA5 wave32 layouts, CDNA5 ISA 7.12.2)
//   A (16x32 f16): lane<16 -> row m=lane,    K = {0..7, 16..23}
//                  lane>=16 -> row m=lane-16, K = {8..15, 24..31}
//   B (32x16 f16): lane<16 -> col n=lane,    K = {0..7, 16..23}
//                  lane>=16 -> col n=lane-16, K = {8..15, 24..31}
//   C/D (16x16 f32, 8 VGPRs): vgpr i, lane<16 -> (M=i,  N=lane)
//                             lane>=16        -> (M=i+8,N=lane-16)
// We stage LDS in fragment order so each lane reads one aligned 32B vector.
// ---------------------------------------------------------------------------
__device__ inline int a_lane(int m, int kloc) { return (kloc & 8)  ? m + 16 : m; }
__device__ inline int a_elem(int kloc)        { return ((kloc & 16) ? 8 : 0) + (kloc & 7); }
__device__ inline int b_lane(int n, int kloc) { return (kloc & 8)  ? n + 16 : n; }
__device__ inline int b_elem(int kloc)        { return ((kloc & 16) ? 8 : 0) + (kloc & 7); }

// ---------------------------------------------------------------------------
// Kernel 1: embedding gather + masked sum over J, id token at s==0
// ---------------------------------------------------------------------------
__global__ void embed_kernel(const int* __restrict__ cat_arr,
                             const int* __restrict__ dt_arr,
                             const int* __restrict__ amount_arr,
                             const int* __restrict__ id_arr,
                             const float* __restrict__ id_table,
                             const float* __restrict__ cat_table,
                             const float* __restrict__ amount_table,
                             const float* __restrict__ dt_table,
                             float* __restrict__ x) {
    int idx = blockIdx.x;
    int b = idx / SEQ;
    int s = idx % SEQ;
    int e = threadIdx.x;

    if (s == 0) {
        x[(size_t)idx * EDIM + e] = id_table[(size_t)id_arr[b] * EDIM + e];
        return;
    }
    int l = s - 1;
    const int* cp = cat_arr    + ((size_t)(b * LSEQ + l)) * JDIM;
    const int* ap = amount_arr + ((size_t)(b * LSEQ + l)) * JDIM;
    float acc = dt_table[(size_t)dt_arr[b * LSEQ + l] * EDIM + e];
    for (int j = 0; j < JDIM; ++j) {
        int c = cp[j];
        if (c != NCAT) {
            acc += cat_table[(size_t)c * EDIM + e];
            acc += amount_table[(size_t)ap[j] * EDIM + e];
        }
    }
    x[(size_t)idx * EDIM + e] = acc;
}

// ---------------------------------------------------------------------------
// Kernel 2: row-panel WMMA GEMM, K=N=EDIM=128.
//   out = epilogue(A[M,128] @ W[128,128] + bias [, residual, LN])
//   mode 0: +bias    mode 1: +bias,relu    mode 2: LN(residual + A@W + bias)
// grid = M/16 blocks, 256 threads (8 wave32, one 16x16 n-tile each).
// A-panel staged into LDS by the Tensor Data Mover (wave 0), tracked with
// TENSORcnt; weights converted straight into fragment-ordered LDS.
// ---------------------------------------------------------------------------
__global__ void gemm_rowpanel_kernel(const float* __restrict__ A,
                                     const float* __restrict__ W,
                                     const float* __restrict__ bias,
                                     const float* __restrict__ residual,
                                     const float* __restrict__ ln_g,
                                     const float* __restrict__ ln_b,
                                     float* __restrict__ out,
                                     int mode) {
    __shared__ __align__(16) float    AsStage[16 * EDIM];       // 8 KB (TDM dest)
    __shared__ __align__(32) _Float16 Af[4][32][16];            // 4 KB
    __shared__ __align__(32) _Float16 Bf[4][8][32][16];         // 32 KB
    __shared__ float    Cs[16][EDIM];                           // 8 KB
    __shared__ float    redA[16][16], redB[16][16];
    __shared__ float    mu_s[16], rs_s[16];

    int tid  = threadIdx.x;
    int wave = tid >> 5;
    int lane = tid & 31;
    int m0   = blockIdx.x * 16;

#if __has_builtin(__builtin_amdgcn_tensor_load_to_lds) && \
    __has_builtin(__builtin_amdgcn_s_wait_tensorcnt)
    // --- TDM: DMA the 16x128 f32 A-panel into LDS (one issue from wave 0) ---
    if (tid < 32) {
        unsigned long long gaddr = (unsigned long long)(const void*)(A + (size_t)m0 * EDIM);
        unsigned           laddr = (unsigned)(unsigned long long)(const void*)AsStage;
        u32x4 g0;
        g0[0] = 1u;                                   // count=1 (valid), user mode
        g0[1] = laddr;                                // lds_addr
        g0[2] = (unsigned)gaddr;                      // global_addr[31:0]
        g0[3] = (unsigned)((gaddr >> 32) & 0x1FFFFFFu) | (2u << 30); // addr msbs | type=2
        i32x8 g1;
        g1[0] = (int)(2u << 16);                      // data_size = 4B
        g1[1] = (int)(128u << 16);                    // tensor_dim0 = 128 (bits 79:48 lo)
        g1[2] = (int)(16u  << 16);                    // tensor_dim1 = 16  (bits 111:80 lo)
        g1[3] = (int)(128u << 16);                    // tile_dim0 = 128   (bits 127:112)
        g1[4] = 16;                                   // tile_dim1 = 16, tile_dim2 = 0
        g1[5] = 128;                                  // tensor_dim0_stride = 128
        g1[6] = 0;
        g1[7] = 0;
        i32x4 z4 = {0, 0, 0, 0};
#if defined(__clang_major__) && (__clang_major__ >= 23)
        i32x8 z8 = {0, 0, 0, 0, 0, 0, 0, 0};
        __builtin_amdgcn_tensor_load_to_lds(g0, g1, z4, z4, z8, 0);
#else
        __builtin_amdgcn_tensor_load_to_lds(g0, g1, z4, z4, 0);
#endif
        __builtin_amdgcn_s_wait_tensorcnt(0);
    }
    __syncthreads();
    // convert staged f32 panel -> fragment-ordered f16
    for (int i = tid; i < 16 * EDIM; i += 256) {
        int m = i >> 7, k = i & 127;
        int kk = k >> 5, kloc = k & 31;
        Af[kk][a_lane(m, kloc)][a_elem(kloc)] = (_Float16)AsStage[i];
    }
#else
    for (int i = tid; i < 16 * EDIM; i += 256) {
        int m = i >> 7, k = i & 127;
        int kk = k >> 5, kloc = k & 31;
        Af[kk][a_lane(m, kloc)][a_elem(kloc)] =
            (_Float16)A[(size_t)(m0 + m) * EDIM + k];
    }
#endif
    // weights -> fragment-ordered f16
    for (int i = tid; i < EDIM * EDIM; i += 256) {
        int k = i >> 7, n = i & 127;
        int kk = k >> 5, kloc = k & 31;
        Bf[kk][n >> 4][b_lane(n & 15, kloc)][b_elem(kloc)] = (_Float16)W[i];
    }
    __syncthreads();

    v8f c = {};
#pragma unroll
    for (int kk = 0; kk < 4; ++kk) {
        v16h a = *(const v16h*)(&Af[kk][lane][0]);       // 2x ds_load_b128
        v16h b = *(const v16h*)(&Bf[kk][wave][lane][0]); // 2x ds_load_b128
        c = __builtin_amdgcn_wmma_f32_16x16x32_f16(
                false, a, false, b, (short)0, c, false, false);
    }

#pragma unroll
    for (int i = 0; i < 8; ++i) {
        int r = (lane < 16) ? i : i + 8;
        Cs[r][wave * 16 + (lane & 15)] = c[i];
    }
    __syncthreads();

    if (mode != 2) {
        for (int i = tid; i < 16 * EDIM; i += 256) {
            int r = i / EDIM, n = i % EDIM;
            float v = Cs[r][n] + bias[n];
            if (mode == 1) v = fmaxf(v, 0.0f);
            out[(size_t)(m0 + r) * EDIM + n] = v;
        }
        return;
    }

    // mode 2: y = residual + C + bias; out = LN(y)*g + b
    for (int i = tid; i < 16 * EDIM; i += 256) {
        int r = i / EDIM, n = i % EDIM;
        Cs[r][n] += bias[n] + residual[(size_t)(m0 + r) * EDIM + n];
    }
    __syncthreads();
    {   // per-row mean / var: 16 threads per row, 8 elems each
        int row = tid >> 4, slot = tid & 15;
        float s = 0.f, s2 = 0.f;
#pragma unroll
        for (int q = 0; q < 8; ++q) {
            float v = Cs[row][slot * 8 + q];
            s += v; s2 += v * v;
        }
        redA[row][slot] = s;
        redB[row][slot] = s2;
    }
    __syncthreads();
    if ((tid & 15) == 0) {
        int row = tid >> 4;
        float s = 0.f, s2 = 0.f;
#pragma unroll
        for (int q = 0; q < 16; ++q) { s += redA[row][q]; s2 += redB[row][q]; }
        float mu  = s * (1.0f / EDIM);
        float var = s2 * (1.0f / EDIM) - mu * mu;
        mu_s[row] = mu;
        rs_s[row] = rsqrtf(var + LN_EPS);
    }
    __syncthreads();
    for (int i = tid; i < 16 * EDIM; i += 256) {
        int r = i / EDIM, n = i % EDIM;
        float v = (Cs[r][n] - mu_s[r]) * rs_s[r] * ln_g[n] + ln_b[n];
        out[(size_t)(m0 + r) * EDIM + n] = v;
    }
}

// ---------------------------------------------------------------------------
// Kernel 3: attention (per (b, h, query) block). S=129, hd=64.
// ---------------------------------------------------------------------------
__global__ void attn_kernel(const float* __restrict__ q,
                            const float* __restrict__ k,
                            const float* __restrict__ v,
                            float* __restrict__ o) {
    int idx = blockIdx.x;
    int si = idx % SEQ;
    int bh = idx / SEQ;
    int h  = bh % HEADS;
    int b  = bh / HEADS;
    int tid = threadIdx.x;

    __shared__ float sc[SEQ];
    __shared__ float qrow[HD];
    __shared__ float mx, sm;

    const float* qp = q + ((size_t)(b * SEQ + si)) * EDIM + h * HD;
    if (tid < HD) qrow[tid] = qp[tid];
    __syncthreads();

    for (int j = tid; j < SEQ; j += blockDim.x) {
        const float* kp = k + ((size_t)(b * SEQ + j)) * EDIM + h * HD;
        float d = 0.f;
#pragma unroll 8
        for (int e = 0; e < HD; ++e) d += qrow[e] * kp[e];
        sc[j] = d * 0.125f;                 // 1/sqrt(64)
    }
    __syncthreads();
    if (tid == 0) {
        float m = -1e30f;
        for (int j = 0; j < SEQ; ++j) m = fmaxf(m, sc[j]);
        mx = m;
    }
    __syncthreads();
    for (int j = tid; j < SEQ; j += blockDim.x) sc[j] = __expf(sc[j] - mx);
    __syncthreads();
    if (tid == 0) {
        float s = 0.f;
        for (int j = 0; j < SEQ; ++j) s += sc[j];
        sm = s;
    }
    __syncthreads();
    float inv = 1.0f / sm;
    for (int d = tid; d < HD; d += blockDim.x) {
        float acc = 0.f;
        for (int j = 0; j < SEQ; ++j)
            acc += sc[j] * v[((size_t)(b * SEQ + j)) * EDIM + h * HD + d];
        o[((size_t)(b * SEQ + si)) * EDIM + h * HD + d] = acc * inv;
    }
}

// ---------------------------------------------------------------------------
// Kernel 4: mean-pool tokens 1..128 -> h[B, E]
// ---------------------------------------------------------------------------
__global__ void pool_kernel(const float* __restrict__ x2, float* __restrict__ h) {
    int b = blockIdx.x, e = threadIdx.x;
    float s = 0.f;
    for (int l = 1; l < SEQ; ++l)
        s += x2[((size_t)(b * SEQ + l)) * EDIM + e];
    h[(size_t)b * EDIM + e] = s * (1.0f / (float)LSEQ);
}

// ---------------------------------------------------------------------------
// Kernel 5: generic N-tiled WMMA GEMM  out = act(A[M,K] @ W[K,N] + bias)
// block -> 16 x 128 tile of C, K loop in steps of 32 with zero padding.
// ---------------------------------------------------------------------------
__global__ void gemm_ntile_kernel(const float* __restrict__ A,
                                  const float* __restrict__ W,
                                  const float* __restrict__ bias,
                                  float* __restrict__ out,
                                  int M, int K, int N, int do_relu) {
    __shared__ __align__(32) _Float16 Af[32][16];       // 1 KB
    __shared__ __align__(32) _Float16 Bf[8][32][16];    // 8 KB

    int tid  = threadIdx.x;
    int wave = tid >> 5;
    int lane = tid & 31;
    int m0   = blockIdx.x * 16;
    int n0   = blockIdx.y * 128;

    v8f c = {};
    int ksteps = (K + 31) / 32;
    for (int kk = 0; kk < ksteps; ++kk) {
        int k0 = kk * 32;
        // prefetch next weight tile into cache hierarchy (global_prefetch_b8)
        if (kk + 1 < ksteps)
            __builtin_prefetch(&W[(size_t)(k0 + 32) * N + n0 + (tid & 127)], 0, 3);
        for (int i = tid; i < 16 * 32; i += 256) {
            int m = i >> 5, kloc = i & 31;
            float v = (k0 + kloc < K) ? A[(size_t)(m0 + m) * K + k0 + kloc] : 0.0f;
            Af[a_lane(m, kloc)][a_elem(kloc)] = (_Float16)v;
        }
        for (int i = tid; i < 32 * 128; i += 256) {
            int kloc = i >> 7, n = i & 127;
            float v = (k0 + kloc < K && n0 + n < N)
                          ? W[(size_t)(k0 + kloc) * N + n0 + n] : 0.0f;
            Bf[n >> 4][b_lane(n & 15, kloc)][b_elem(kloc)] = (_Float16)v;
        }
        __syncthreads();
        v16h a = *(const v16h*)(&Af[lane][0]);
        v16h b = *(const v16h*)(&Bf[wave][lane][0]);
        c = __builtin_amdgcn_wmma_f32_16x16x32_f16(
                false, a, false, b, (short)0, c, false, false);
        __syncthreads();
    }

#pragma unroll
    for (int i = 0; i < 8; ++i) {
        int r   = (lane < 16) ? i : i + 8;
        int col = n0 + wave * 16 + (lane & 15);
        if (col < N && m0 + r < M) {
            float v = c[i] + bias[col];
            if (do_relu) v = fmaxf(v, 0.0f);
            out[(size_t)(m0 + r) * N + col] = v;
        }
    }
}

// ---------------------------------------------------------------------------
// Host-side orchestration
// ---------------------------------------------------------------------------
extern "C" void kernel_launch(void* const* d_in, const int* in_sizes, int n_in,
                              void* d_out, int out_size, void* d_ws, size_t ws_size,
                              hipStream_t stream) {
    const int*   cat_arr      = (const int*)  d_in[0];
    const int*   dt_arr       = (const int*)  d_in[1];
    const int*   amount_arr   = (const int*)  d_in[2];
    const int*   id_arr       = (const int*)  d_in[3];
    const float* id_table     = (const float*)d_in[4];
    const float* cat_table    = (const float*)d_in[5];
    const float* amount_table = (const float*)d_in[6];
    const float* dt_table     = (const float*)d_in[7];
    const float* Wq = (const float*)d_in[8],  *bq = (const float*)d_in[9];
    const float* Wk = (const float*)d_in[10], *bk = (const float*)d_in[11];
    const float* Wv = (const float*)d_in[12], *bv = (const float*)d_in[13];
    const float* Wo = (const float*)d_in[14], *bo = (const float*)d_in[15];
    const float* ln1_g = (const float*)d_in[16], *ln1_b = (const float*)d_in[17];
    const float* W1 = (const float*)d_in[18], *b1 = (const float*)d_in[19];
    const float* W2 = (const float*)d_in[20], *b2 = (const float*)d_in[21];
    const float* ln2_g = (const float*)d_in[22], *ln2_b = (const float*)d_in[23];
    const float* lin1_W = (const float*)d_in[24], *lin1_b = (const float*)d_in[25];
    const float* lin2_W = (const float*)d_in[26], *lin2_b = (const float*)d_in[27];

    const size_t M    = (size_t)BATCH * SEQ;       // 16512 rows
    const size_t NBIG = M * EDIM;                  // floats per [B,S,E] buffer

    float* ws  = (float*)d_ws;
    float* x   = ws;                // embeddings / residual 1
    float* qb  = ws + 1 * NBIG;     // Q     -> later x1 (post-LN1)
    float* kb_ = ws + 2 * NBIG;     // K     -> later ffn mid t
    float* vb  = ws + 3 * NBIG;     // V     -> later x2 (post-LN2)
    float* ob  = ws + 4 * NBIG;     // attn out -> later h / x1p
    float* hb  = ob;                         // [128,128]
    float* x1p = ob + BATCH * EDIM;          // [128,1000]

    (void)in_sizes; (void)n_in; (void)out_size; (void)ws_size;

    const int mtiles = (int)(M / 16);        // 1032

    // 1. embeddings
    embed_kernel<<<BATCH * SEQ, EDIM, 0, stream>>>(
        cat_arr, dt_arr, amount_arr, id_arr,
        id_table, cat_table, amount_table, dt_table, x);

    // 2. Q, K, V projections
    gemm_rowpanel_kernel<<<mtiles, 256, 0, stream>>>(x, Wq, bq, nullptr, nullptr, nullptr, qb, 0);
    gemm_rowpanel_kernel<<<mtiles, 256, 0, stream>>>(x, Wk, bk, nullptr, nullptr, nullptr, kb_, 0);
    gemm_rowpanel_kernel<<<mtiles, 256, 0, stream>>>(x, Wv, bv, nullptr, nullptr, nullptr, vb, 0);

    // 3. attention
    attn_kernel<<<BATCH * HEADS * SEQ, 128, 0, stream>>>(qb, kb_, vb, ob);

    // 4. output projection + residual + LN1  -> x1 (reuse qb)
    float* x1 = qb;
    gemm_rowpanel_kernel<<<mtiles, 256, 0, stream>>>(ob, Wo, bo, x, ln1_g, ln1_b, x1, 2);

    // 5. FFN: t = relu(x1@W1+b1) (reuse kb_); x2 = LN2(x1 + t@W2+b2) (reuse vb)
    float* t  = kb_;
    float* x2 = vb;
    gemm_rowpanel_kernel<<<mtiles, 256, 0, stream>>>(x1, W1, b1, nullptr, nullptr, nullptr, t, 1);
    gemm_rowpanel_kernel<<<mtiles, 256, 0, stream>>>(t, W2, b2, x1, ln2_g, ln2_b, x2, 2);

    // 6. mean pool tokens 1..128
    pool_kernel<<<BATCH, EDIM, 0, stream>>>(x2, hb);

    // 7. classifier head
    dim3 g1(BATCH / 16, (NCAT + 127) / 128);
    gemm_ntile_kernel<<<g1, 256, 0, stream>>>(hb, lin1_W, lin1_b, x1p,
                                              BATCH, EDIM, NCAT, 1);
    gemm_ntile_kernel<<<g1, 256, 0, stream>>>(x1p, lin2_W, lin2_b, (float*)d_out,
                                              BATCH, NCAT, NCAT, 0);
}
